// SwinTransformerModel_46308337386119
// MI455X (gfx1250) — compile-verified, matrix operand order checked by reading, not run on hardware
//
#include <hip/hip_runtime.h>
#include <math.h>

// ---------------------------------------------------------------------------
// Types for CDNA5 WMMA (wave32): V_WMMA_F32_16X16X32_BF16
// ---------------------------------------------------------------------------
typedef __attribute__((ext_vector_type(16))) __bf16 v16bf;
typedef __attribute__((ext_vector_type(8)))  __bf16 v8bf;
typedef __attribute__((ext_vector_type(4)))  __bf16 v4bf;
typedef __attribute__((ext_vector_type(8)))  float  v8f;

#define AS_STRIDE 40  // bf16 per LDS row: 80B => 16B-aligned rows, conflict-free frag reads

// 16x32 bf16 fragment from a row-major LDS tile (stride AS_STRIDE), per the
// CDNA5 wave32 layout: lane l: m=l&15, h=l>>4; elems 0..7 <- K=h*8+0..7,
// elems 8..15 <- K=16+h*8+0..7. Both halves are contiguous 16B runs.
__device__ __forceinline__ v16bf load_frag(const __bf16* rowbase, int lane) {
    int m = lane & 15, hh = lane >> 4;
    const __bf16* p = rowbase + m * AS_STRIDE + hh * 8;
    v8bf lo = *(const v8bf*)(p);
    v8bf hi = *(const v8bf*)(p + 16);
    v16bf r;
#pragma unroll
    for (int e = 0; e < 8; ++e) { r[e] = lo[e]; r[e + 8] = hi[e]; }
    return r;
}

// ---------------------------------------------------------------------------
// GEMM: C[M,N] = op(A[M,K] @ W[N,K]^T + bias) (+= if ACC)
// flags: 1 = bias, 2 = exact GELU, 4 = accumulate (residual)
// Requires M%64==0, N%32==0, K%32==0 (true for every GEMM in this network).
// 128 threads = 4 waves; block tile 64x32; wave (wm,wn) computes 32x16 via two
// WMMAs sharing one B fragment. LDS double-buffered, float4 global loads
// issued one K-step ahead, single __syncthreads per K-step. fp32->bf16 uses
// plain casts so the backend selects hardware packed converts.
// ---------------------------------------------------------------------------
__global__ void __launch_bounds__(128)
gemm_bf16_wmma_kernel(const float* __restrict__ A, const float* __restrict__ W,
                      const float* __restrict__ bias, float* __restrict__ C_,
                      int M, int N, int K, int flags) {
    __shared__ alignas(16) __bf16 As[2][64 * AS_STRIDE];
    __shared__ alignas(16) __bf16 Bs[2][32 * AS_STRIDE];
    const int m0 = blockIdx.y * 64, n0 = blockIdx.x * 32;
    const int tid = threadIdx.x;
    const int wave = tid >> 5, lane = tid & 31;
    const int wm = wave >> 1, wn = wave & 1;

    float4 ra[4], rb[2];
    auto load_tile = [&](int k0) {
#pragma unroll
        for (int i = 0; i < 4; ++i) {           // A: 64x32 = 512 float4 groups
            int idx = i * 128 + tid, r = idx >> 3, c = (idx & 7) * 4;
            ra[i] = *(const float4*)&A[(size_t)(m0 + r) * K + k0 + c];
        }
#pragma unroll
        for (int i = 0; i < 2; ++i) {           // B: 32x32 = 256 float4 groups
            int idx = i * 128 + tid, r = idx >> 3, c = (idx & 7) * 4;
            rb[i] = *(const float4*)&W[(size_t)(n0 + r) * K + k0 + c];
        }
    };
    auto store_tile = [&](int b) {
#pragma unroll
        for (int i = 0; i < 4; ++i) {
            int idx = i * 128 + tid, r = idx >> 3, c = (idx & 7) * 4;
            v4bf t = { (__bf16)ra[i].x, (__bf16)ra[i].y, (__bf16)ra[i].z, (__bf16)ra[i].w };
            *(v4bf*)&As[b][r * AS_STRIDE + c] = t;
        }
#pragma unroll
        for (int i = 0; i < 2; ++i) {
            int idx = i * 128 + tid, r = idx >> 3, c = (idx & 7) * 4;
            v4bf t = { (__bf16)rb[i].x, (__bf16)rb[i].y, (__bf16)rb[i].z, (__bf16)rb[i].w };
            *(v4bf*)&Bs[b][r * AS_STRIDE + c] = t;
        }
    };

    v8f acc0 = {}, acc1 = {};
    const int nk = K >> 5;
    load_tile(0);
    for (int i = 0; i < nk; ++i) {
        const int buf = i & 1;
        store_tile(buf);
        __syncthreads();
        if (i + 1 < nk) load_tile((i + 1) << 5);
        v16bf a0 = load_frag(&As[buf][(wm * 32) * AS_STRIDE], lane);
        v16bf a1 = load_frag(&As[buf][(wm * 32 + 16) * AS_STRIDE], lane);
        v16bf bf = load_frag(&Bs[buf][(wn * 16) * AS_STRIDE], lane);
        acc0 = __builtin_amdgcn_wmma_f32_16x16x32_bf16(false, a0, false, bf,
                                                       (short)0, acc0, false, false);
        acc1 = __builtin_amdgcn_wmma_f32_16x16x32_bf16(false, a1, false, bf,
                                                       (short)0, acc1, false, false);
    }

    // Epilogue: D lane layout: col = lane&15; rows = 8*(lane>>4) + v
    const int col = n0 + wn * 16 + (lane & 15);
    const int hh = lane >> 4;
    const float bv = (flags & 1) ? bias[col] : 0.f;
#pragma unroll
    for (int v = 0; v < 8; ++v) {
        int row = m0 + wm * 32 + hh * 8 + v;
        float val = acc0[v] + bv;
        if (flags & 2) val = 0.5f * val * (1.f + erff(val * 0.70710678118654752f));
        size_t idx = (size_t)row * N + col;
        if (flags & 4) C_[idx] += val; else C_[idx] = val;

        int row2 = row + 16;
        float val2 = acc1[v] + bv;
        if (flags & 2) val2 = 0.5f * val2 * (1.f + erff(val2 * 0.70710678118654752f));
        size_t idx2 = (size_t)row2 * N + col;
        if (flags & 4) C_[idx2] += val2; else C_[idx2] = val2;
    }
}

// ---------------------------------------------------------------------------
// LayerNorm over last dim C (C <= 3072). One block (128 thr) per row.
// Safe in-place (each element owned by exactly one thread, cached in regs).
// ---------------------------------------------------------------------------
__global__ void __launch_bounds__(128)
ln_kernel(const float* __restrict__ in, float* __restrict__ out,
          const float* __restrict__ g, const float* __restrict__ b, int C) {
    const size_t row = blockIdx.x;
    const int t = threadIdx.x;
    const float* x = in + row * C;
    float vals[24];
    int nv = 0; float s = 0.f;
    for (int c = t; c < C; c += 128) { float v = x[c]; vals[nv++] = v; s += v; }
    __shared__ float red[128];
    red[t] = s; __syncthreads();
    for (int o = 64; o > 0; o >>= 1) { if (t < o) red[t] += red[t + o]; __syncthreads(); }
    float mean = red[0] / C; __syncthreads();
    s = 0.f;
    for (int i = 0; i < nv; ++i) { float d = vals[i] - mean; s += d * d; }
    red[t] = s; __syncthreads();
    for (int o = 64; o > 0; o >>= 1) { if (t < o) red[t] += red[t + o]; __syncthreads(); }
    float rstd = rsqrtf(red[0] / C + 1e-5f);
    float* y = out + row * C;
    nv = 0;
    for (int c = t; c < C; c += 128) y[c] = (vals[nv++] - mean) * rstd * g[c] + b[c];
}

// ---------------------------------------------------------------------------
// Patch embed: Conv2d(3,96,k=4,s=4) NCHW -> NHWC tokens. One block per pixel.
// ---------------------------------------------------------------------------
__global__ void __launch_bounds__(96)
patch_embed_kernel(const float* __restrict__ x, const float* __restrict__ w,
                   const float* __restrict__ bias, float* __restrict__ out) {
    const int pix = blockIdx.x, oc = threadIdx.x;
    const int b = pix / 3136, r = pix % 3136, h = r / 56, ww = r % 56;
    float s = bias[oc];
    for (int c = 0; c < 3; ++c)
        for (int kh = 0; kh < 4; ++kh)
            for (int kw = 0; kw < 4; ++kw)
                s += x[((size_t)(b * 3 + c) * 224 + (h * 4 + kh)) * 224 + (ww * 4 + kw)] *
                     w[((oc * 3 + c) * 4 + kh) * 4 + kw];
    out[(size_t)pix * 96 + oc] = s;
}

// ---------------------------------------------------------------------------
// Window attention (7x7, head_dim=32). One 64-thread block per (b,window,head).
// Cyclic shift handled via rolled-coordinate indexing; shift mask groups
// computed from rolled coords; relative position bias from window geometry.
// qkv buffer: [tokens, 3*C] with columns ((which*heads)+head)*32 + d.
// ---------------------------------------------------------------------------
__global__ void __launch_bounds__(64)
attn_kernel(const float* __restrict__ qkv, const float* __restrict__ rpb,
            float* __restrict__ out, int Hs, int Ws, int C, int heads, int shift) {
    const int nWx = Ws / 7, nwin = (Hs / 7) * nWx;
    const int bid = blockIdx.x;
    const int head = bid % heads;
    const int t1 = bid / heads;
    const int win = t1 % nwin, b = t1 / nwin;
    const int wy = win / nWx, wx = win % nWx;
    __shared__ float qs[49 * 32], ks[49 * 32], vs[49 * 32];
    __shared__ int gid[49], tok[49];
    const int tid = threadIdx.x;
    const int C3 = 3 * C;

    if (tid < 49) {
        int i = tid / 7, j = tid % 7;
        int hr = wy * 7 + i, wr = wx * 7 + j;
        int h0 = hr, w0 = wr;
        if (shift) { h0 = (hr + shift) % Hs; w0 = (wr + shift) % Ws; }
        tok[tid] = (b * Hs + h0) * Ws + w0;
        int rh = (hr < Hs - 7) ? 0 : ((hr < Hs - shift) ? 1 : 2);
        int rw = (wr < Ws - 7) ? 0 : ((wr < Ws - shift) ? 1 : 2);
        gid[tid] = shift ? (rh * 3 + rw) : 0;
    }
    __syncthreads();

    const float qscale = 0.17677669529663687f;  // 32^-0.5
    for (int e = tid; e < 49 * 32; e += 64) {
        int r = e >> 5, d = e & 31;
        size_t base = (size_t)tok[r] * C3;
        qs[e] = qkv[base + (size_t)head * 32 + d] * qscale;
        ks[e] = qkv[base + (size_t)(heads + head) * 32 + d];
        vs[e] = qkv[base + (size_t)(2 * heads + head) * 32 + d];
    }
    __syncthreads();

    if (tid < 49) {
        const int i = tid, ri = i / 7, ci = i % 7, gi = gid[i];
        float s[49], mx = -1e30f;
        for (int j = 0; j < 49; ++j) {
            float d = 0.f;
#pragma unroll 8
            for (int kk = 0; kk < 32; ++kk) d += qs[i * 32 + kk] * ks[j * 32 + kk];
            int rj = j / 7, cj = j % 7;
            int rel = (ri - rj + 6) * 13 + (ci - cj + 6);
            d += rpb[rel * heads + head];
            if (shift && gi != gid[j]) d -= 100.f;
            s[j] = d; mx = fmaxf(mx, d);
        }
        float sum = 0.f;
        for (int j = 0; j < 49; ++j) { float e2 = expf(s[j] - mx); s[j] = e2; sum += e2; }
        const float inv = 1.f / sum;
        float acc[32];
#pragma unroll
        for (int d = 0; d < 32; ++d) acc[d] = 0.f;
        for (int j = 0; j < 49; ++j) {
            float p = s[j] * inv;
#pragma unroll
            for (int d = 0; d < 32; ++d) acc[d] += p * vs[j * 32 + d];
        }
        size_t obase = (size_t)tok[i] * C + (size_t)head * 32;
#pragma unroll
        for (int d = 0; d < 32; ++d) out[obase + d] = acc[d];
    }
}

// ---------------------------------------------------------------------------
// Patch-merge gather: X[B,Hs,Ws,C] -> Hm[B*H2*W2, 4C], concat order
// [(0,0),(1,0),(0,1),(1,1)] matching the reference.
// ---------------------------------------------------------------------------
__global__ void merge_gather_kernel(const float* __restrict__ X, float* __restrict__ Hm,
                                    int Hs, int Ws, int C, size_t total) {
    size_t idx = (size_t)blockIdx.x * 256 + threadIdx.x;
    if (idx >= total) return;
    const int H2 = Hs / 2, W2 = Ws / 2;
    int c = (int)(idx % C);
    size_t r1 = idx / C;
    int q = (int)(r1 % 4);
    size_t row = r1 / 4;
    int wv = (int)(row % W2);
    size_t r2 = row / W2;
    int hv = (int)(r2 % H2);
    int b = (int)(r2 / H2);
    int dh = (q == 1 || q == 3) ? 1 : 0;
    int dw = (q >= 2) ? 1 : 0;
    Hm[idx] = X[(((size_t)b * Hs + (2 * hv + dh)) * Ws + (2 * wv + dw)) * C + c];
}

__global__ void __launch_bounds__(256)
pool_kernel(const float* __restrict__ Hin, float* __restrict__ pooled) {
    const int b = blockIdx.x;
    for (int c = threadIdx.x; c < 768; c += 256) {
        float s = 0.f;
        for (int t = 0; t < 49; ++t) s += Hin[((size_t)b * 49 + t) * 768 + c];
        pooled[b * 768 + c] = s * (1.f / 49.f);
    }
}

__global__ void __launch_bounds__(128)
head_kernel(const float* __restrict__ pooled, const float* __restrict__ w,
            const float* __restrict__ bias, float* __restrict__ out) {
    const int b = blockIdx.x, o = threadIdx.x;
    if (o >= 100) return;
    float s = bias[o];
    for (int c = 0; c < 768; ++c) s += pooled[b * 768 + c] * w[o * 768 + c];
    out[b * 100 + o] = s;
}

// ---------------------------------------------------------------------------
// Host orchestration
// ---------------------------------------------------------------------------
// Input flattening (JAX pytree order: dict keys sorted alphabetically):
//   params: head_b, head_w,
//           merges[0..2]: {norm_b, norm_g, red_w},
//           norm0_b, norm0_g, normf_b, normf_g, patch_b, patch_w,
//           stages[0..3].blocks[*]: {fc1_b, fc1_w, fc2_b, fc2_w, n1_b, n1_g,
//                                    n2_b, n2_g, proj_b, proj_w, qkv_b, qkv_w, rpb}
//   then x ('x' > 'params'). Launch also tolerates x-first ordering.
static const int DEPTHS_[4] = {2, 2, 6, 2};
enum { L_FC1_B = 0, L_FC1_W, L_FC2_B, L_FC2_W, L_N1_B, L_N1_G, L_N2_B, L_N2_G,
       L_PROJ_B, L_PROJ_W, L_QKV_B, L_QKV_W, L_RPB };

extern "C" void kernel_launch(void* const* d_in, const int* in_sizes, int n_in,
                              void* d_out, int out_size, void* d_ws, size_t ws_size,
                              hipStream_t stream) {
    const int NX = 64 * 3 * 224 * 224;
    int po = 0, xi = n_in - 1;
    if (in_sizes[0] == NX) { po = 1; xi = 0; }  // x flattened first instead of last
    auto P = [&](int i) { return (const float*)d_in[i + po]; };
    const float* xin = (const float*)d_in[xi];

    // Workspace: X (current activations), H (LN / attn-out scratch), S (qkv/mlp)
    char* ws = (char*)d_ws;
    float* X  = (float*)ws;                          // 200704*96*4  = 77,070,336 B
    float* Hb = (float*)(ws + 77070336);             // 77,070,336 B
    float* S  = (float*)(ws + 154140672);            // 200704*384*4 = 308,281,344 B

    // --- patch embed + norm0 ---
    patch_embed_kernel<<<200704, 96, 0, stream>>>(xin, P(16), P(15), X);
    ln_kernel<<<200704, 128, 0, stream>>>(X, X, P(12), P(11), 96);

    int Hs = 56, Wd = 56, C = 96, heads = 3, cum = 0;
    for (int s = 0; s < 4; ++s) {
        const int M = 64 * Hs * Wd;
        for (int blk = 0; blk < DEPTHS_[s]; ++blk) {
            const int L = 17 + (cum + blk) * 13;
            const int shift = (blk % 2 == 1 && Hs > 7) ? 3 : 0;
            // attention branch
            ln_kernel<<<M, 128, 0, stream>>>(X, Hb, P(L + L_N1_G), P(L + L_N1_B), C);
            gemm_bf16_wmma_kernel<<<dim3(3 * C / 32, M / 64), 128, 0, stream>>>(
                Hb, P(L + L_QKV_W), P(L + L_QKV_B), S, M, 3 * C, C, 1);
            const int nwin = (Hs / 7) * (Wd / 7);
            attn_kernel<<<64 * nwin * heads, 64, 0, stream>>>(
                S, P(L + L_RPB), Hb, Hs, Wd, C, heads, shift);
            gemm_bf16_wmma_kernel<<<dim3(C / 32, M / 64), 128, 0, stream>>>(
                Hb, P(L + L_PROJ_W), P(L + L_PROJ_B), X, M, C, C, 1 | 4);
            // MLP branch
            ln_kernel<<<M, 128, 0, stream>>>(X, Hb, P(L + L_N2_G), P(L + L_N2_B), C);
            gemm_bf16_wmma_kernel<<<dim3(4 * C / 32, M / 64), 128, 0, stream>>>(
                Hb, P(L + L_FC1_W), P(L + L_FC1_B), S, M, 4 * C, C, 1 | 2);
            gemm_bf16_wmma_kernel<<<dim3(C / 32, M / 64), 128, 0, stream>>>(
                S, P(L + L_FC2_W), P(L + L_FC2_B), X, M, C, 4 * C, 1 | 4);
        }
        cum += DEPTHS_[s];
        if (s < 3) {
            const int H2 = Hs / 2, W2 = Wd / 2, M2 = 64 * H2 * W2;
            const size_t tot = (size_t)M2 * 4 * C;
            merge_gather_kernel<<<(unsigned)((tot + 255) / 256), 256, 0, stream>>>(
                X, Hb, Hs, Wd, C, tot);
            ln_kernel<<<M2, 128, 0, stream>>>(Hb, Hb, P(3 + 3 * s), P(2 + 3 * s), 4 * C);
            gemm_bf16_wmma_kernel<<<dim3(2 * C / 32, M2 / 64), 128, 0, stream>>>(
                Hb, P(4 + 3 * s), nullptr, X, M2, 2 * C, 4 * C, 0);
            Hs = H2; Wd = W2; C *= 2; heads *= 2;
        }
    }

    // --- final norm, pool, classification head ---
    ln_kernel<<<3136, 128, 0, stream>>>(X, Hb, P(14), P(13), 768);
    pool_kernel<<<64, 256, 0, stream>>>(Hb, S);
    head_kernel<<<64, 128, 0, stream>>>(S, P(1), P(0), (float*)d_out);
}
